// FrequencyBandAttention_13649406067368
// MI455X (gfx1250) — matrix operand order repaired and checked by reading, not compiled, not run on hardware
//
#include <hip/hip_runtime.h>
#include <hip/hip_bf16.h>
#include <math.h>

typedef __attribute__((ext_vector_type(16))) __bf16 v16bf;
typedef __attribute__((ext_vector_type(8)))  float  v8f;

#define D_MODEL 512
#define N_HEADS 8
#define DK      64
#define PATCH   64
#define NBANDS  5

// ---------- bf16 helpers ----------
__device__ __forceinline__ unsigned f2bf_bits(float f) {
    unsigned u = __float_as_uint(f);
    unsigned r = (u + 0x7FFFu + ((u >> 16) & 1u)) >> 16;   // RNE
    return r & 0xFFFFu;
}

union FragBF { v16bf v; unsigned u[8]; };
union FragF  { v8f   v; float    f[8]; };

// A-matrix (16x32 bf16) K index base for VGPR v / wave-half h (contiguous order: k = 16*h + 2*v)
__device__ __forceinline__ int a_kbase(int v, int half) {
    return 2 * v + 16 * half;
}

__device__ __forceinline__ v8f wmma_bf16(const FragBF& a, const FragBF& b, v8f c) {
    return __builtin_amdgcn_wmma_f32_16x16x32_bf16(false, a.v, false, b.v,
                                                   (short)0, c, false, false);
}

// ---------- CDNA5 async global->LDS copy (ASYNCcnt-tracked) ----------
__device__ __forceinline__ void async_copy_b128(unsigned lds_off, unsigned long long ga) {
    asm volatile("global_load_async_to_lds_b128 %0, %1, off"
                 :: "v"(lds_off), "v"(ga) : "memory");
}
__device__ __forceinline__ void wait_async_16() {
    asm volatile("s_wait_asynccnt 0x10" ::: "memory");
}
__device__ __forceinline__ void wait_async_0() {
    asm volatile("s_wait_asynccnt 0x0" ::: "memory");
}

// ---------- f32 -> bf16 bulk convert ----------
__global__ void cvt_f32_bf16(const float* __restrict__ in, unsigned short* __restrict__ out, int n) {
    int i = blockIdx.x * blockDim.x + threadIdx.x;
    if (i < n) out[i] = (unsigned short)f2bf_bits(in[i]);
}

// ---------- band power -> per-head additive bias (B,H,N) f32 ----------
__global__ void band_bias_kernel(const float* __restrict__ patches,
                                 const float* __restrict__ band_bias, // (H, 5)
                                 float* __restrict__ biasout,         // (B, H, N)
                                 int BN, int Nseq) {
    int idx = blockIdx.x * blockDim.x + threadIdx.x;
    if (idx >= BN) return;
    float p[PATCH];
    const float* src = patches + (size_t)idx * PATCH;
#pragma unroll
    for (int t = 0; t < PATCH; ++t) p[t] = src[t];

    float pw[26];
    const float w0 = 2.0f * 3.14159265358979323846f / (float)PATCH;
    for (int k = 1; k <= 25; ++k) {
        float re = 0.f, im = 0.f;
#pragma unroll
        for (int t = 0; t < PATCH; ++t) {
            float ang = w0 * (float)(k * t);
            re += p[t] * __cosf(ang);
            im += p[t] * __sinf(ang);
        }
        pw[k] = re * re + im * im;
    }
    // sorted band names: alpha, beta, delta, gamma, theta
    float bp[NBANDS];
    bp[0] = pw[3];
    bp[1] = 0.25f * (pw[4] + pw[5] + pw[6] + pw[7]);
    bp[2] = pw[1];
    float g = 0.f;
    for (int k = 8; k <= 25; ++k) g += pw[k];
    bp[3] = g / 18.0f;
    bp[4] = pw[2];
#pragma unroll
    for (int c = 0; c < NBANDS; ++c) bp[c] = log1pf(bp[c]);

    int b = idx / Nseq, n = idx % Nseq;
#pragma unroll
    for (int h = 0; h < N_HEADS; ++h) {
        float acc = 0.f;
#pragma unroll
        for (int c = 0; c < NBANDS; ++c) acc += bp[c] * band_bias[h * NBANDS + c];
        biasout[((size_t)b * N_HEADS + h) * Nseq + n] = acc;
    }
}

// ---------- bf16 WMMA GEMM, register-blocked 16x64 per wave ----------
// out = A(rows,K) @ W(ncols,K)^T + bias
// mode 0: store bf16 to (B,H,N,64)   (Q/K layout)
// mode 1: store bf16 to (B,H,64,N)   (V transposed layout)
// mode 2: store f32  to (rows,ncols) (final output)
__global__ void wmma_gemm(const unsigned short* __restrict__ A,
                          const unsigned short* __restrict__ Wt,
                          const float* __restrict__ bvec,
                          unsigned short* __restrict__ outbf,
                          float* __restrict__ outf,
                          int K, int ncols, int mode, int Nseq) {
    const int lane = threadIdx.x & 31;
    const int wid  = threadIdx.x >> 5;          // 4 waves / block
    const int half = lane >> 4;
    const int l16  = lane & 15;
    const int tn0    = (blockIdx.x * 4 + wid) * 4;  // first 16-col subtile index
    const int tile_m = blockIdx.y;

    const unsigned short* ar = A + (size_t)(tile_m * 16 + l16) * K;
    const unsigned short* br[4];
#pragma unroll
    for (int t = 0; t < 4; ++t)
        br[t] = Wt + (size_t)((tn0 + t) * 16 + l16) * K;

    FragF c[4];
#pragma unroll
    for (int t = 0; t < 4; ++t)
#pragma unroll
        for (int r = 0; r < 8; ++r) c[t].f[r] = 0.f;

    const int ksteps = K / 32;
    for (int ks = 0; ks < ksteps; ++ks) {
        const int k0 = ks * 32;
        FragBF af;                                  // A loaded once, reused 4x
#pragma unroll
        for (int v = 0; v < 8; ++v)
            af.u[v] = *(const unsigned*)(ar + k0 + a_kbase(v, half));
#pragma unroll
        for (int t = 0; t < 4; ++t) {
            FragBF bf;
#pragma unroll
            for (int v = 0; v < 8; ++v)
                bf.u[v] = *(const unsigned*)(br[t] + k0 + half * 16 + 2 * v);
            c[t].v = wmma_bf16(af, bf, c[t].v);
        }
    }

#pragma unroll
    for (int t = 0; t < 4; ++t) {
        const int col = (tn0 + t) * 16 + l16;
        const float bb = bvec ? bvec[col] : 0.f;
#pragma unroll
        for (int r = 0; r < 8; ++r) {
            int row = tile_m * 16 + r + 8 * half;
            float val = c[t].f[r] + bb;
            if (mode == 2) {
                outf[(size_t)row * ncols + col] = val;
            } else {
                int b = row / Nseq, n = row % Nseq;
                int h = col >> 6, d = col & 63;
                size_t base = (size_t)(b * N_HEADS + h);
                size_t off = (mode == 0) ? (base * Nseq + n) * DK + d
                                         : (base * DK + d) * (size_t)Nseq + n;
                outbf[off] = (unsigned short)f2bf_bits(val);
            }
        }
    }
}

// ---------- flash attention: one wave per (b,h,16-query tile) ----------
// K/V tiles double-buffered in LDS via global_load_async_to_lds_b128.
__global__ void attn_kernel(const unsigned short* __restrict__ Qb, // (B,H,N,64) bf16
                            const unsigned short* __restrict__ Kb, // (B,H,N,64) bf16
                            const unsigned short* __restrict__ Vt, // (B,H,64,N) bf16
                            const float* __restrict__ bias,        // (B,H,N)
                            unsigned short* __restrict__ ctx,      // (B*N,512) bf16
                            int Nseq) {
    const int lane = threadIdx.x & 31;
    const int half = lane >> 4;
    const int l16  = lane & 15;

    const int qtiles = Nseq / 16;
    int bid = blockIdx.x;
    const int qt = bid % qtiles;  bid /= qtiles;
    const int h  = bid % N_HEADS;
    const int b  = bid / N_HEADS;

    const size_t headoff = (size_t)(b * N_HEADS + h) * Nseq * DK;
    const unsigned short* Qh = Qb + headoff;
    const unsigned short* Kh = Kb + headoff;
    const unsigned short* Vh = Vt + (size_t)(b * N_HEADS + h) * DK * Nseq;
    const float* biash = bias + (size_t)(b * N_HEADS + h) * Nseq;

    __shared__ unsigned short KL[2][32][DK];   // 32 keys x 64 d      (4KB per buffer)
    __shared__ unsigned short VL[2][DK][32];   // 64 d x 32 keys      (4KB per buffer)
    __shared__ unsigned short ldsP[16][32];

    // Q A-fragments over d_k (two 32-wide K slices)
    FragBF qf[2];
    {
        const unsigned short* qr = Qh + (size_t)(qt * 16 + l16) * DK;
#pragma unroll
        for (int ks = 0; ks < 2; ++ks)
#pragma unroll
            for (int v = 0; v < 8; ++v)
                qf[ks].u[v] = *(const unsigned*)(qr + ks * 32 + a_kbase(v, half));
    }

    float bq[8];
#pragma unroll
    for (int r = 0; r < 8; ++r) bq[r] = biash[qt * 16 + r + 8 * half];

    float mrow[8], lrow[8];
    FragF oc[4];
#pragma unroll
    for (int r = 0; r < 8; ++r) { mrow[r] = -1e30f; lrow[r] = 0.f; }
#pragma unroll
    for (int t = 0; t < 4; ++t)
#pragma unroll
        for (int r = 0; r < 8; ++r) oc[t].f[r] = 0.f;

    // ---- async stage of one 32-key chunk into LDS buffer `buf` ----
    auto stage = [&](int kb, int buf) {
        // K chunk: rows kb..kb+31 are contiguous (4KB) in global memory
        unsigned klbase = (unsigned)(size_t)&KL[buf][0][0];
        unsigned long long kg = (unsigned long long)(size_t)(Kh + (size_t)kb * DK);
#pragma unroll
        for (int i = 0; i < 8; ++i) {
            unsigned off = (unsigned)(i * 512 + lane * 16);
            async_copy_b128(klbase + off, kg + off);
        }
        // V chunk: 64 rows of 64B, global row stride = Nseq*2 bytes
        unsigned vlbase = (unsigned)(size_t)&VL[buf][0][0];
        unsigned long long vg = (unsigned long long)(size_t)(Vh + kb);
        int vrow = lane >> 2;            // 0..7
        int vcol = (lane & 3) * 16;      // byte within 64B row
#pragma unroll
        for (int i = 0; i < 8; ++i) {
            int row = i * 8 + vrow;
            async_copy_b128(vlbase + (unsigned)(row * 64 + vcol),
                            vg + (unsigned long long)row * (unsigned)(Nseq * 2) + (unsigned)vcol);
        }
    };

    const float scale = 0.125f; // 1/sqrt(64)
    const int nch = Nseq / 32;

    stage(0, 0);
    for (int cix = 0; cix < nch; ++cix) {
        const int buf = cix & 1;
        if (cix + 1 < nch) { stage((cix + 1) * 32, buf ^ 1); wait_async_16(); }
        else               { wait_async_0(); }
        const int kb = cix * 32;

        // ---- scores from LDS K tile: two 16x16 fragments ----
        FragF s[2];
#pragma unroll
        for (int sub = 0; sub < 2; ++sub) {
            const unsigned short* kr = &KL[buf][sub * 16 + l16][0];
            FragBF b0, b1;
#pragma unroll
            for (int v = 0; v < 8; ++v) {
                b0.u[v] = *(const unsigned*)(kr + half * 16 + 2 * v);
                b1.u[v] = *(const unsigned*)(kr + 32 + half * 16 + 2 * v);
            }
            v8f acc = {};
            acc = wmma_bf16(qf[0], b0, acc);
            acc = wmma_bf16(qf[1], b1, acc);
            s[sub].v = acc;
        }
        const float bk0 = biash[kb + l16];
        const float bk1 = biash[kb + 16 + l16];

        // ---- online softmax over this 32-key chunk ----
        float p0[8], p1[8];
#pragma unroll
        for (int r = 0; r < 8; ++r) {
            float s0 = s[0].f[r] * scale + bq[r] + bk0;
            float s1 = s[1].f[r] * scale + bq[r] + bk1;
            float mx = fmaxf(s0, s1);
#pragma unroll
            for (int off = 1; off < 16; off <<= 1)
                mx = fmaxf(mx, __shfl_xor(mx, off, 32));
            float nm = fmaxf(mrow[r], mx);
            float corr = __expf(mrow[r] - nm);
            mrow[r] = nm;
            p0[r] = __expf(s0 - nm);
            p1[r] = __expf(s1 - nm);
            float rs = p0[r] + p1[r];
#pragma unroll
            for (int off = 1; off < 16; off <<= 1)
                rs += __shfl_xor(rs, off, 32);
            lrow[r] = lrow[r] * corr + rs;
#pragma unroll
            for (int t = 0; t < 4; ++t) oc[t].f[r] *= corr;
        }

        // ---- relayout P (D-layout -> A-layout) through LDS ----
        __syncthreads();
#pragma unroll
        for (int r = 0; r < 8; ++r) {
            int m = r + 8 * half;
            ldsP[m][l16]      = (unsigned short)f2bf_bits(p0[r]);
            ldsP[m][16 + l16] = (unsigned short)f2bf_bits(p1[r]);
        }
        __syncthreads();
        FragBF pa;
        {
            const unsigned short* pr = &ldsP[l16][0];
#pragma unroll
            for (int v = 0; v < 8; ++v)
                pa.u[v] = *(const unsigned*)(pr + a_kbase(v, half));
        }

        // ---- O += P @ V from LDS V tile ----
#pragma unroll
        for (int t = 0; t < 4; ++t) {
            const unsigned short* vr = &VL[buf][t * 16 + l16][0];
            FragBF vb;
#pragma unroll
            for (int v = 0; v < 8; ++v)
                vb.u[v] = *(const unsigned*)(vr + half * 16 + 2 * v);
            oc[t].v = wmma_bf16(pa, vb, oc[t].v);
        }
    }

    // ---- epilogue: normalize, write bf16 context (B*N, 512) ----
#pragma unroll
    for (int r = 0; r < 8; ++r) {
        float inv = 1.0f / lrow[r];
        size_t row = (size_t)b * Nseq + qt * 16 + r + 8 * half;
#pragma unroll
        for (int t = 0; t < 4; ++t) {
            int col = h * DK + t * 16 + l16;
            ctx[row * D_MODEL + col] = (unsigned short)f2bf_bits(oc[t].f[r] * inv);
        }
    }
}

extern "C" void kernel_launch(void* const* d_in, const int* in_sizes, int n_in,
                              void* d_out, int out_size, void* d_ws, size_t ws_size,
                              hipStream_t stream) {
    const float* x        = (const float*)d_in[0];
    const float* patches  = (const float*)d_in[1];
    const float* W_q      = (const float*)d_in[2];
    const float* b_q      = (const float*)d_in[3];
    const float* W_k      = (const float*)d_in[4];
    const float* b_k      = (const float*)d_in[5];
    const float* W_v      = (const float*)d_in[6];
    const float* b_v      = (const float*)d_in[7];
    const float* W_o      = (const float*)d_in[8];
    const float* b_o      = (const float*)d_in[9];
    const float* band_bias= (const float*)d_in[10];
    float* out = (float*)d_out;

    const int BN   = in_sizes[0] / D_MODEL;   // B*N = 4096
    const int Nseq = 2048;
    const int B    = BN / Nseq;

    // workspace carve-up (bf16 stored as ushort)
    char* ws = (char*)d_ws;
    size_t off = 0;
    auto carve = [&](size_t bytes) { void* p = ws + off; off += (bytes + 255) & ~(size_t)255; return p; };
    unsigned short* xb  = (unsigned short*)carve((size_t)BN * D_MODEL * 2);
    unsigned short* wqb = (unsigned short*)carve((size_t)D_MODEL * D_MODEL * 2);
    unsigned short* wkb = (unsigned short*)carve((size_t)D_MODEL * D_MODEL * 2);
    unsigned short* wvb = (unsigned short*)carve((size_t)D_MODEL * D_MODEL * 2);
    unsigned short* wob = (unsigned short*)carve((size_t)D_MODEL * D_MODEL * 2);
    unsigned short* Qb  = (unsigned short*)carve((size_t)BN * D_MODEL * 2);  // (B,H,N,64)
    unsigned short* Kb  = (unsigned short*)carve((size_t)BN * D_MODEL * 2);  // (B,H,N,64)
    unsigned short* Vt  = (unsigned short*)carve((size_t)BN * D_MODEL * 2);  // (B,H,64,N)
    unsigned short* ctx = (unsigned short*)carve((size_t)BN * D_MODEL * 2);  // (B*N,512)
    float* biasbuf      = (float*)carve((size_t)B * N_HEADS * Nseq * 4);
    (void)ws_size; (void)n_in; (void)out_size;

    // 1) convert inputs/weights to bf16
    {
        int n = BN * D_MODEL;
        cvt_f32_bf16<<<(n + 255) / 256, 256, 0, stream>>>(x, xb, n);
        int nw = D_MODEL * D_MODEL;
        cvt_f32_bf16<<<(nw + 255) / 256, 256, 0, stream>>>(W_q, wqb, nw);
        cvt_f32_bf16<<<(nw + 255) / 256, 256, 0, stream>>>(W_k, wkb, nw);
        cvt_f32_bf16<<<(nw + 255) / 256, 256, 0, stream>>>(W_v, wvb, nw);
        cvt_f32_bf16<<<(nw + 255) / 256, 256, 0, stream>>>(W_o, wob, nw);
    }

    // 2) frequency-band bias
    band_bias_kernel<<<(BN + 127) / 128, 128, 0, stream>>>(patches, band_bias, biasbuf, BN, Nseq);

    // 3) Q/K/V projections (bf16 WMMA GEMMs, 16x64 per wave)
    {
        dim3 grid(D_MODEL / 256, BN / 16);   // 4 waves/block, each wave a 16x64 tile
        wmma_gemm<<<grid, 128, 0, stream>>>(xb, wqb, b_q, Qb, nullptr, D_MODEL, D_MODEL, 0, Nseq);
        wmma_gemm<<<grid, 128, 0, stream>>>(xb, wkb, b_k, Kb, nullptr, D_MODEL, D_MODEL, 0, Nseq);
        wmma_gemm<<<grid, 128, 0, stream>>>(xb, wvb, b_v, Vt, nullptr, D_MODEL, D_MODEL, 1, Nseq);
    }

    // 4) flash attention with band bias (async double-buffered K/V tiles)
    attn_kernel<<<B * N_HEADS * (Nseq / 16), 32, 0, stream>>>(Qb, Kb, Vt, biasbuf, ctx, Nseq);

    // 5) output projection (f32 store straight to d_out)
    {
        dim3 grid(D_MODEL / 256, BN / 16);
        wmma_gemm<<<grid, 128, 0, stream>>>(ctx, wob, b_o, nullptr, out, D_MODEL, D_MODEL, 2, Nseq);
    }
}
